// SimpleVectorQuantizer_89773406421457
// MI455X (gfx1250) — compile-verified, matrix operand order checked by reading, not run on hardware
//
#include <hip/hip_runtime.h>

// ---------------------------------------------------------------------------
// SimpleVectorQuantizer for MI455X (gfx1250, wave32, WMMA + TDM)
// Distances via split-bf16 WMMA:  z·c ≈ zhi·chi + zhi·clo + zlo·chi  (f32 acc)
// Codebook panels staged into LDS by the Tensor Data Mover (double-buffered).
// ---------------------------------------------------------------------------

typedef __attribute__((ext_vector_type(16))) __bf16 v16bf;
typedef __attribute__((ext_vector_type(8)))  __bf16 v8bf;
typedef __attribute__((ext_vector_type(8)))  float  v8f;
typedef __attribute__((ext_vector_type(4)))  unsigned int v4u;
typedef __attribute__((ext_vector_type(8)))  int  v8i;
typedef __attribute__((ext_vector_type(4)))  int  v4i;

#define NVEC 65536      // B*H*W
#define DDIM 128
#define KCODE 1024
#define HW 1024         // H*W
#define NB 64           // batch
#define TOT_ZD 8388608  // NVEC*DDIM

#define EMA_DECAY 0.99f
#define EMA_REST  0.01f
#define EPS 1e-5f

// TDM chunking of the codebook panel
#define CHUNK_TILES 2                    // 16-code tiles per chunk
#define CHUNK_CODES (CHUNK_TILES * 16)   // 32 codes
#define CHUNK_ELEMS (CHUNK_CODES * DDIM) // 4096 bf16 = 8 KB
#define NCHUNKS (KCODE / CHUNK_CODES)    // 32

// ---- output layout (flat f32, reference return order) ----
#define OFF_Q    0u
#define OFF_IDX  8388608u
#define OFF_CL   8454144u
#define OFF_CBL  8454145u
#define OFF_NCB  8454146u
#define OFF_NEC  8585218u
#define OFF_NEW  8586242u

// ---- workspace layout (byte offsets) ----
#define WS_ZHI     0u          // N*D bf16  = 16 MiB
#define WS_ZLO     16777216u   // N*D bf16
#define WS_Z2      33554432u   // N f32
#define WS_CBHI    33816576u   // K*D bf16
#define WS_CBLO    34078720u   // K*D bf16
#define WS_C2      34340864u   // K f32
#define WS_IDX     34344960u   // N int32
#define WS_COUNTS  34607104u   // K f32        } contiguous zero region
#define WS_DW      34611200u   // K*D f32      }
#define WS_SCAL    35135488u   // [0]=loss_sum [1]=n_sum  }
#define ZERO_FLOATS 132098     // counts + dw + 2 scalars

__device__ __forceinline__ v16bf load_bf16_operand(const __bf16* base) {
  // elements 0..7  = base[0..7]   (K = half*8 + 0..7)
  // elements 8..15 = base[16..23] (K = half*8 + 16..23)
  v8bf a = *reinterpret_cast<const v8bf*>(base);
  v8bf b = *reinterpret_cast<const v8bf*>(base + 16);
  v16bf r;
#pragma unroll
  for (int i = 0; i < 8; ++i) { r[i] = a[i]; r[i + 8] = b[i]; }
  return r;
}

__device__ __forceinline__ void atomic_add_f32(float* p, float v) {
  __hip_atomic_fetch_add(p, v, __ATOMIC_RELAXED, __HIP_MEMORY_SCOPE_AGENT);
}

// Issue a 1-D TDM copy: n8 8-byte elements, global -> LDS (byte offset lds_off).
// D# group0: count=1|type=2, lds_addr, 57-bit global addr.
// D# group1: data_size=3 (8B), tensor_dim0 = tile_dim0 = n8, stride = n8.
// 6-arg builtin form (clang-23 / therock headers): g0, g1, g2, g3, g4, cpol.
__device__ __forceinline__ void tdm_load_1d(const void* gptr, unsigned int lds_off,
                                            unsigned int n8) {
  const unsigned long long ga = (unsigned long long)(uintptr_t)gptr;
  v4u g0;
  g0.x = 1u;                                                   // count=1 (valid)
  g0.y = lds_off;                                              // LDS byte address
  g0.z = (unsigned int)ga;                                     // global addr [31:0]
  g0.w = (unsigned int)((ga >> 32) & 0x1FFFFFFu) | (2u << 30); // addr[56:32] | type=2
  v8i g1 = {};
  g1[0] = (int)(3u << 16);                   // data_size = 8 bytes
  g1[1] = (int)((n8 & 0xFFFFu) << 16);       // tensor_dim0[15:0]  (bits 63:48)
  g1[2] = (int)(n8 >> 16);                   // tensor_dim0[31:16] (bits 79:64)
  g1[3] = (int)((n8 & 0xFFFFu) << 16);       // tile_dim0          (bits 127:112)
  g1[5] = (int)n8;                           // tensor_dim0_stride[31:0]
  v4i gz4 = {};
  v8i gz8 = {};
  __builtin_amdgcn_tensor_load_to_lds(g0, g1, gz4, gz4, gz8, 0);
}

// ---------------------------------------------------------------------------
__global__ void vq_zero(float* p, int n) {
  int i = blockIdx.x * blockDim.x + threadIdx.x;
  int stride = gridDim.x * blockDim.x;
  for (; i < n; i += stride) p[i] = 0.0f;
}

// z [B,C,H,W] f32  ->  zhi/zlo [N,D] bf16 (split), z2 [N] row norms
__global__ void vq_prep_z(const float* __restrict__ z,
                          __bf16* __restrict__ zhi, __bf16* __restrict__ zlo,
                          float* __restrict__ z2) {
  const int b = blockIdx.x;        // 64
  const int hw = threadIdx.x;      // 1024
  const int n = b * HW + hw;
  float s = 0.0f;
#pragma unroll 4
  for (int c = 0; c < DDIM; ++c) {
    float v = z[(size_t)(b * DDIM + c) * HW + hw];   // coalesced over hw
    __bf16 h = (__bf16)v;
    zhi[(size_t)n * DDIM + c] = h;
    zlo[(size_t)n * DDIM + c] = (__bf16)(v - (float)h);
    s += v * v;
  }
  z2[n] = s;
}

// codebook [K,D] f32 -> cbhi/cblo bf16 split + c2 norms
__global__ void vq_prep_cb(const float* __restrict__ cb,
                           __bf16* __restrict__ cbhi, __bf16* __restrict__ cblo,
                           float* __restrict__ c2) {
  __shared__ float sm[DDIM];
  const int k = blockIdx.x, d = threadIdx.x;
  float v = cb[(size_t)k * DDIM + d];
  __bf16 h = (__bf16)v;
  cbhi[(size_t)k * DDIM + d] = h;
  cblo[(size_t)k * DDIM + d] = (__bf16)(v - (float)h);
  sm[d] = v * v;
  __syncthreads();
#pragma unroll
  for (int o = DDIM / 2; o > 0; o >>= 1) {
    if (d < o) sm[d] += sm[d + o];
    __syncthreads();
  }
  if (d == 0) c2[k] = sm[0];
}

// ---------------------------------------------------------------------------
// GEMM + argmin.  One wave per 16-row tile of z (4096 tiles, 8 waves/block).
// Codebook panel double-buffered in LDS via TDM; 12 WMMAs per code tile.
// __launch_bounds__(256, 1): allocate VGPRs for low occupancy — latency is
// hidden by the TDM double-buffer + WMMA chains, and this kills the scratch
// spill of the hoisted A operands seen at default budgets.
// ---------------------------------------------------------------------------
__global__ void __launch_bounds__(256, 1)
vq_gemm_argmin(const __bf16* __restrict__ zhi,
               const __bf16* __restrict__ zlo,
               const float*  __restrict__ z2,
               const __bf16* __restrict__ cbhi,
               const __bf16* __restrict__ cblo,
               const float*  __restrict__ c2,
               int*   __restrict__ idx_out,
               float* __restrict__ idx_f_out) {
  // [buf][ hi panel | lo panel ]  -> 2 * 2 * 8KB = 32 KB static LDS
  __shared__ __align__(16) __bf16 ldsb[2][2 * CHUNK_ELEMS];

  const int lane = threadIdx.x & 31;
  const int wave = threadIdx.x >> 5;
  const int tile = blockIdx.x * 8 + wave;  // 0..4095
  const int col  = lane & 15;              // B-column / C-column this lane owns
  const int half = lane >> 4;              // K-group selector

  // Hoist A (this wave's 16 z rows, all 128 K) into registers: 8 x v16bf.
  const int rowA = tile * 16 + col;        // A: lane L holds row M = L%16
  v16bf Ahi[4], Alo[4];
#pragma unroll
  for (int c = 0; c < 4; ++c) {
    const size_t off = (size_t)rowA * DDIM + c * 32 + half * 8;
    Ahi[c] = load_bf16_operand(zhi + off);
    Alo[c] = load_bf16_operand(zlo + off);
  }
  float z2v[8];
#pragma unroll
  for (int r = 0; r < 8; ++r) z2v[r] = z2[tile * 16 + half * 8 + r];

  float minv[8];
  int   mini[8];
#pragma unroll
  for (int r = 0; r < 8; ++r) { minv[r] = 3.0e38f; mini[r] = 0; }

  const unsigned int lds_base = (unsigned int)(uintptr_t)&ldsb[0][0];
  const unsigned int buf_bytes = 2u * CHUNK_ELEMS * sizeof(__bf16); // 16 KB
  const unsigned int panel8 = (CHUNK_ELEMS * 2) / 8;                // 8B elems

  // Prologue: TDM-stage chunk 0 into buffer 0 (wave 0 only; uniform branch).
  if (wave == 0) {
    tdm_load_1d(cbhi, lds_base, panel8);
    tdm_load_1d(cblo, lds_base + CHUNK_ELEMS * 2, panel8);
  }

  for (int ch = 0; ch < NCHUNKS; ++ch) {
    if (wave == 0) __builtin_amdgcn_s_wait_tensorcnt(0);  // chunk ch resident
    __syncthreads();
    if (wave == 0 && ch + 1 < NCHUNKS) {
      const unsigned int nb = (unsigned int)((ch + 1) & 1) * buf_bytes;
      tdm_load_1d(cbhi + (size_t)(ch + 1) * CHUNK_ELEMS, lds_base + nb, panel8);
      tdm_load_1d(cblo + (size_t)(ch + 1) * CHUNK_ELEMS,
                  lds_base + nb + CHUNK_ELEMS * 2, panel8);
    }

    const __bf16* Lhi = &ldsb[ch & 1][0];
    const __bf16* Llo = &ldsb[ch & 1][CHUNK_ELEMS];

#pragma unroll
    for (int t = 0; t < CHUNK_TILES; ++t) {
      const int code = (ch * CHUNK_TILES + t) * 16 + col;  // lane's column code
      const int cl   = t * 16 + col;                       // code within chunk
      v8f acc = {};
#pragma unroll
      for (int c = 0; c < 4; ++c) {
        const int off = cl * DDIM + c * 32 + half * 8;
        v16bf Bhi = load_bf16_operand(Lhi + off);
        v16bf Blo = load_bf16_operand(Llo + off);
        acc = __builtin_amdgcn_wmma_f32_16x16x32_bf16(false, Ahi[c], false, Bhi,
                                                      (short)0, acc, false, false);
        acc = __builtin_amdgcn_wmma_f32_16x16x32_bf16(false, Ahi[c], false, Blo,
                                                      (short)0, acc, false, false);
        acc = __builtin_amdgcn_wmma_f32_16x16x32_bf16(false, Alo[c], false, Bhi,
                                                      (short)0, acc, false, false);
      }
      const float c2v = c2[code];
#pragma unroll
      for (int r = 0; r < 8; ++r) {
        // C layout: VGPR r -> row (half*8 + r), column col
        float dist = z2v[r] + c2v - 2.0f * acc[r];
        if (dist < minv[r]) { minv[r] = dist; mini[r] = code; }  // first-min kept
      }
    }
  }

  // Reduce across the 16 columns (lanes with same `half`): butterfly in-half.
#pragma unroll
  for (int r = 0; r < 8; ++r) {
#pragma unroll
    for (int m = 8; m >= 1; m >>= 1) {
      float ov = __shfl_xor(minv[r], m, 32);
      int   oi = __shfl_xor(mini[r], m, 32);
      if (ov < minv[r] || (ov == minv[r] && oi < mini[r])) {
        minv[r] = ov; mini[r] = oi;
      }
    }
    if (col == 0) {
      const int row = tile * 16 + half * 8 + r;
      idx_out[row]   = mini[r];
      idx_f_out[row] = (float)mini[r];
    }
  }
}

// ---------------------------------------------------------------------------
// Gather codebook rows -> quantized output; EMA segment sums via L2 atomics.
__global__ void vq_gather_stats(const float* __restrict__ z,
                                const float* __restrict__ cb,
                                const int*   __restrict__ idx,
                                float* __restrict__ out_q,
                                float* __restrict__ counts,
                                float* __restrict__ dw) {
  const int n = blockIdx.x;    // 65536
  const int d = threadIdx.x;   // 128
  const int k = idx[n];
  out_q[(size_t)n * DDIM + d] = cb[(size_t)k * DDIM + d];
  const int b = n >> 10, hw = n & (HW - 1);
  const float zv = z[(size_t)(b * DDIM + d) * HW + hw];
  atomic_add_f32(&dw[(size_t)k * DDIM + d], zv);
  if (d == 0) atomic_add_f32(&counts[k], 1.0f);
}

// mean((quantized - z)^2) over the flat 8.4M buffers (faithful raw-reinterpret)
__global__ void vq_loss(const float* __restrict__ q, const float* __restrict__ z,
                        float* __restrict__ loss_sum) {
  __shared__ float sm[256];
  float s = 0.0f;
  size_t i = (size_t)blockIdx.x * blockDim.x + threadIdx.x;
  const size_t stride = (size_t)gridDim.x * blockDim.x;
  for (; i < (size_t)TOT_ZD; i += stride) {
    float dlt = q[i] - z[i];
    s += dlt * dlt;
  }
  sm[threadIdx.x] = s;
  __syncthreads();
#pragma unroll
  for (int o = 128; o > 0; o >>= 1) {
    if (threadIdx.x < (unsigned)o) sm[threadIdx.x] += sm[threadIdx.x + o];
    __syncthreads();
  }
  if (threadIdx.x == 0) atomic_add_f32(loss_sum, sm[0]);
}

// new_ema_count + its total (single 1024-thread block)
__global__ void vq_ema_count(const float* __restrict__ ec,
                             const float* __restrict__ counts,
                             float* __restrict__ nec_out,
                             float* __restrict__ n_sum) {
  __shared__ float sm[KCODE];
  const int k = threadIdx.x;
  float nec = EMA_DECAY * ec[k] + EMA_REST * counts[k];
  nec_out[k] = nec;
  sm[k] = nec;
  __syncthreads();
#pragma unroll
  for (int o = KCODE / 2; o > 0; o >>= 1) {
    if (k < o) sm[k] += sm[k + o];
    __syncthreads();
  }
  if (k == 0) *n_sum = sm[0];
}

__global__ void vq_ema_weight(const float* __restrict__ ew,
                              const float* __restrict__ dw,
                              const float* __restrict__ nec,
                              const float* __restrict__ n_sum,
                              float* __restrict__ new_cb,
                              float* __restrict__ new_ew) {
  const int k = blockIdx.x, d = threadIdx.x;
  const float n = *n_sum;
  const float w = (nec[k] + EPS) / (n + (float)KCODE * EPS) * n;
  const float e = EMA_DECAY * ew[(size_t)k * DDIM + d] + EMA_REST * dw[(size_t)k * DDIM + d];
  new_ew[(size_t)k * DDIM + d] = e;
  new_cb[(size_t)k * DDIM + d] = e / w;
}

__global__ void vq_finalize_loss(const float* __restrict__ loss_sum,
                                 float* __restrict__ out_cl,
                                 float* __restrict__ out_cbl) {
  const float m = *loss_sum * (1.0f / (float)TOT_ZD);
  *out_cl = m;   // commitment loss
  *out_cbl = m;  // codebook loss (same value; stop_gradient is identity fwd)
}

// ---------------------------------------------------------------------------
extern "C" void kernel_launch(void* const* d_in, const int* in_sizes, int n_in,
                              void* d_out, int out_size, void* d_ws, size_t ws_size,
                              hipStream_t stream) {
  const float* z  = (const float*)d_in[0];
  const float* cb = (const float*)d_in[1];
  const float* ec = (const float*)d_in[2];
  const float* ew = (const float*)d_in[3];

  float* out = (float*)d_out;
  char* ws = (char*)d_ws;

  __bf16* zhi   = (__bf16*)(ws + WS_ZHI);
  __bf16* zlo   = (__bf16*)(ws + WS_ZLO);
  float*  z2    = (float*)(ws + WS_Z2);
  __bf16* cbhi  = (__bf16*)(ws + WS_CBHI);
  __bf16* cblo  = (__bf16*)(ws + WS_CBLO);
  float*  c2    = (float*)(ws + WS_C2);
  int*    idx   = (int*)(ws + WS_IDX);
  float*  cnts  = (float*)(ws + WS_COUNTS);
  float*  dw    = (float*)(ws + WS_DW);
  float*  scal  = (float*)(ws + WS_SCAL);  // [0]=loss_sum [1]=n_sum

  // 1. zero the accumulator region (counts + dw + scalars) every call
  vq_zero<<<(ZERO_FLOATS + 255) / 256, 256, 0, stream>>>(cnts, ZERO_FLOATS);

  // 2/3. precision split + norms
  vq_prep_z<<<NB, HW, 0, stream>>>(z, zhi, zlo, z2);
  vq_prep_cb<<<KCODE, DDIM, 0, stream>>>(cb, cbhi, cblo, c2);

  // 4. WMMA distance GEMM + argmin (4096 row-tiles, 8 waves/block, TDM staging)
  vq_gemm_argmin<<<NVEC / 16 / 8, 256, 0, stream>>>(zhi, zlo, z2, cbhi, cblo, c2,
                                                    idx, out + OFF_IDX);

  // 5. gather + EMA segment sums
  vq_gather_stats<<<NVEC, DDIM, 0, stream>>>(z, cb, idx, out + OFF_Q, cnts, dw);

  // 6. losses
  vq_loss<<<1024, 256, 0, stream>>>(out + OFF_Q, z, &scal[0]);

  // 7. EMA finalize
  vq_ema_count<<<1, KCODE, 0, stream>>>(ec, cnts, out + OFF_NEC, &scal[1]);
  vq_ema_weight<<<KCODE, DDIM, 0, stream>>>(ew, dw, out + OFF_NEC, &scal[1],
                                            out + OFF_NCB, out + OFF_NEW);
  vq_finalize_loss<<<1, 1, 0, stream>>>(&scal[0], out + OFF_CL, out + OFF_CBL);
}
